// NTMCopy_31920196944408
// MI455X (gfx1250) — compile-verified, matrix operand order checked by reading, not run on hardware
//
#include <hip/hip_runtime.h>
#include <hip/hip_bf16.h>
#include <math.h>

// ---------------------------------------------------------------------------
// NTM cell step for MI455X (gfx1250, wave32).
// Dense GEMMs -> v_wmma_f32_16x16x32_f16 with pre-packed f16 operands and
// unconditional 16B fragment loads (two contiguous 8-half runs per lane).
// Memory-tensor passes -> wide coalesced f32 streaming (bandwidth-bound,
// ~400MB HBM traffic => ~20us roofline at 23.3 TB/s).
// ---------------------------------------------------------------------------

typedef __attribute__((ext_vector_type(16))) _Float16 v16h;
typedef __attribute__((ext_vector_type(8)))  _Float16 v8h;
typedef __attribute__((ext_vector_type(8)))  float    v8f;

#define BB 256
#define NN 2048
#define MM 64
#define HH 256
#define NI 8
#define NO 8
#define INCAT 72          // NI + MM
#define KCAT 352          // 8 + 64 + 256 padded to /32 (328 -> 352)
#define G4H 1024          // 4*H

__device__ __forceinline__ float sigf(float x) { return 1.0f / (1.0f + __expf(-x)); }
__device__ __forceinline__ float softplusf(float x) {
    return (x > 20.0f) ? x : log1pf(__expf(x));
}

// ---------------------------------------------------------------------------
// Packing kernels: emit zero-padded row-major f16 operands for the WMMA GEMMs.
// ---------------------------------------------------------------------------
__global__ void pack_inp(const float* __restrict__ x, const float* __restrict__ rd,
                         const float* __restrict__ h, _Float16* __restrict__ out) {
    int idx = blockIdx.x * blockDim.x + threadIdx.x;       // B * KCAT
    if (idx >= BB * KCAT) return;
    int b = idx / KCAT, c = idx % KCAT;
    float v = 0.0f;
    if (c < NI)              v = x[b * NI + c];
    else if (c < INCAT)      v = rd[b * MM + (c - NI)];
    else if (c < INCAT + HH) v = h[b * HH + (c - INCAT)];
    out[idx] = (_Float16)v;
}

__global__ void pack_w(const float* __restrict__ Wih, const float* __restrict__ Whh,
                       _Float16* __restrict__ out) {
    int idx = blockIdx.x * blockDim.x + threadIdx.x;       // 4H * KCAT
    if (idx >= G4H * KCAT) return;
    int j = idx / KCAT, c = idx % KCAT;
    float v = 0.0f;
    if (c < INCAT)           v = Wih[j * INCAT + c];
    else if (c < INCAT + HH) v = Whh[j * HH + (c - INCAT)];
    out[idx] = (_Float16)v;
}

// generic: src [rows x cols] f32 -> dst [drows x cols] f16, zero row padding
__global__ void pack_f16(const float* __restrict__ src, int rows, int cols,
                         _Float16* __restrict__ dst, int drows) {
    int idx = blockIdx.x * blockDim.x + threadIdx.x;
    if (idx >= drows * cols) return;
    int r = idx / cols, c = idx % cols;
    dst[idx] = (_Float16)((r < rows) ? src[r * cols + c] : 0.0f);
}

__global__ void pack_bias(const float* __restrict__ bih, const float* __restrict__ bhh,
                          float* __restrict__ out) {
    int j = blockIdx.x * blockDim.x + threadIdx.x;
    if (j < G4H) out[j] = bih[j] + bhh[j];
}

// ---------------------------------------------------------------------------
// WMMA GEMM:  C[256 x Nrows] = A[256 x K]_f16 * W[Npad x K]_f16^T + bias
// One wave per 16x16 C tile; fragment = two unconditional 16B loads per
// operand per 32-K chunk (K-map runs are contiguous: lanes<16 -> [0,8)+[16,24),
// lanes>=16 -> [8,16)+[24,32)). K multiple of 32, Npad multiple of 16.
// ---------------------------------------------------------------------------
__global__ void wmma_gemm_bias(const _Float16* __restrict__ A, int lda,
                               const _Float16* __restrict__ W, int ldw,
                               const float* __restrict__ bias,
                               float* __restrict__ C, int ldc, int Nrows,
                               int K, int ntilesN) {
    int wave = (blockIdx.x * blockDim.x + threadIdx.x) >> 5;
    int lane = threadIdx.x & 31;
    int mt = wave / ntilesN;
    int nt = wave % ntilesN;
    int half_off = (lane < 16) ? 0 : 8;
    const _Float16* Arow = A + (size_t)(mt * 16 + (lane & 15)) * lda + half_off;
    const _Float16* Wrow = W + (size_t)(nt * 16 + (lane & 15)) * ldw + half_off;

    v8f acc = {};
    for (int k0 = 0; k0 < K; k0 += 32) {
        v8h a0 = *(const v8h*)(Arow + k0);
        v8h a1 = *(const v8h*)(Arow + k0 + 16);
        v8h b0 = *(const v8h*)(Wrow + k0);
        v8h b1 = *(const v8h*)(Wrow + k0 + 16);
        v16h a, b;
#pragma unroll
        for (int i = 0; i < 8; ++i) {
            a[i] = a0[i]; a[i + 8] = a1[i];
            b[i] = b0[i]; b[i + 8] = b1[i];
        }
        acc = __builtin_amdgcn_wmma_f32_16x16x32_f16(
            false, a, false, b, (short)0, acc, false, false);
    }

    int col = nt * 16 + (lane & 15);
    if (col < Nrows) {
        float bb = bias[col];
        int rbase = mt * 16 + ((lane < 16) ? 0 : 8);
#pragma unroll
        for (int g = 0; g < 8; ++g)
            C[(size_t)(rbase + g) * ldc + col] = acc[g] + bb;
    }
}

// ---------------------------------------------------------------------------
// LSTM activations: gates (B x 4H) -> c_new, h_new (f32 out) + h16 (f16 copy)
// ---------------------------------------------------------------------------
__global__ void lstm_act(const float* __restrict__ gates, const float* __restrict__ c,
                         float* __restrict__ c_new, float* __restrict__ h_new,
                         _Float16* __restrict__ h16) {
    int idx = blockIdx.x * blockDim.x + threadIdx.x;       // B*H
    if (idx >= BB * HH) return;
    int b = idx / HH, t = idx % HH;
    const float* g = gates + (size_t)b * G4H;
    float gi = g[t], gf = g[HH + t], gg = g[2 * HH + t], go = g[3 * HH + t];
    float cn = sigf(gf) * c[idx] + sigf(gi) * tanhf(gg);
    float hn = sigf(go) * tanhf(cn);
    c_new[idx] = cn;
    h_new[idx] = hn;
    h16[idx] = (_Float16)hn;
}

// ---------------------------------------------------------------------------
// Head params: beta, g, s(softmax of softplus), gamma, |k+eps|; e/a for write.
// params layout per batch: [beta, g, s0, s1, s2, gamma, knorm, pad]
// ---------------------------------------------------------------------------
__global__ void head_params(const float* __restrict__ head, int ldh,
                            float* __restrict__ params,
                            float* __restrict__ e, float* __restrict__ a,
                            int is_write) {
    int b = blockIdx.x * blockDim.x + threadIdx.x;
    if (b >= BB) return;
    const float* r = head + (size_t)b * ldh;
    float beta = softplusf(r[MM]);
    float gg   = sigf(r[MM + 1]);
    float s0 = softplusf(r[MM + 2]);
    float s1 = softplusf(r[MM + 3]);
    float s2 = softplusf(r[MM + 4]);
    float mx = fmaxf(s0, fmaxf(s1, s2));
    float e0 = __expf(s0 - mx), e1 = __expf(s1 - mx), e2 = __expf(s2 - mx);
    float es = e0 + e1 + e2;
    float gamma = 1.0f + softplusf(r[MM + 5]);
    float nk = 0.0f;
    for (int m = 0; m < MM; ++m) {
        float kv = r[m] + 1e-16f;
        nk += kv * kv;
    }
    nk = sqrtf(nk);
    float* p = params + b * 8;
    p[0] = beta; p[1] = gg; p[2] = e0 / es; p[3] = e1 / es; p[4] = e2 / es;
    p[5] = gamma; p[6] = nk; p[7] = 0.0f;
    if (is_write) {
        for (int m = 0; m < MM; ++m) {
            e[b * MM + m] = sigf(r[MM + 6 + m]);
            a[b * MM + m] = r[2 * MM + 6 + m];
        }
    }
}

// ---------------------------------------------------------------------------
// Cosine similarity * beta for all (b, n). Streams the 134MB memory tensor.
// ---------------------------------------------------------------------------
__global__ void sim_kernel(const float* __restrict__ mem,
                           const float* __restrict__ head, int ldh,
                           const float* __restrict__ params,
                           float* __restrict__ sim) {
    int n = blockIdx.x * blockDim.x + threadIdx.x;
    int b = blockIdx.y;
    if (n >= NN) return;
    const float4* row = (const float4*)(mem + ((size_t)b * NN + n) * MM);
    const float4* k4  = (const float4*)(head + (size_t)b * ldh);
    float dot = 0.0f, rn = 0.0f;
#pragma unroll 4
    for (int m = 0; m < MM / 4; ++m) {
        float4 mv = row[m];
        float4 kv = k4[m];
        float mx = mv.x + 1e-16f, my = mv.y + 1e-16f, mz = mv.z + 1e-16f, mw = mv.w + 1e-16f;
        dot += mx * (kv.x + 1e-16f) + my * (kv.y + 1e-16f)
             + mz * (kv.z + 1e-16f) + mw * (kv.w + 1e-16f);
        rn  += mx * mx + my * my + mz * mz + mw * mw;
    }
    rn = sqrtf(rn);
    float beta = params[b * 8 + 0];
    float nk   = params[b * 8 + 6];
    sim[(size_t)b * NN + n] = beta * dot / (fmaxf(rn, 1e-8f) * fmaxf(nk, 1e-8f));
}

// ---------------------------------------------------------------------------
// Per-batch: softmax over N, interpolate with w_prev, circular shift, sharpen,
// renormalize. One 256-thread block per batch; wg staged in LDS (8KB).
// ---------------------------------------------------------------------------
__global__ void address_finalize(const float* __restrict__ sim,
                                 const float* __restrict__ params,
                                 const float* __restrict__ wprev,
                                 float* __restrict__ wout) {
    __shared__ float wg[NN];
    __shared__ float red[256];
    int b = blockIdx.x, t = threadIdx.x;
    const float* s = sim + (size_t)b * NN;

    float mx = -3.4e38f;
    for (int n = t; n < NN; n += 256) mx = fmaxf(mx, s[n]);
    red[t] = mx; __syncthreads();
    for (int off = 128; off > 0; off >>= 1) {
        if (t < off) red[t] = fmaxf(red[t], red[t + off]);
        __syncthreads();
    }
    float gmax = red[0]; __syncthreads();

    float sum = 0.0f;
    for (int n = t; n < NN; n += 256) {
        float ev = __expf(s[n] - gmax);
        wg[n] = ev;
        sum += ev;
    }
    red[t] = sum; __syncthreads();
    for (int off = 128; off > 0; off >>= 1) {
        if (t < off) red[t] += red[t + off];
        __syncthreads();
    }
    float gsum = red[0]; __syncthreads();

    const float* p = params + b * 8;
    float g = p[1], s0 = p[2], s1 = p[3], s2 = p[4], gamma = p[5];
    for (int n = t; n < NN; n += 256)
        wg[n] = g * (wg[n] / gsum) + (1.0f - g) * wprev[(size_t)b * NN + n];
    __syncthreads();

    float wloc[NN / 256];
    float psum = 0.0f;
#pragma unroll
    for (int j = 0; j < NN / 256; ++j) {
        int n = t + j * 256;
        float wh = s0 * wg[(n + NN - 1) & (NN - 1)] + s1 * wg[n]
                 + s2 * wg[(n + 1) & (NN - 1)];
        float w = __powf(fmaxf(wh, 0.0f), gamma);
        wloc[j] = w;
        psum += w;
    }
    red[t] = psum; __syncthreads();
    for (int off = 128; off > 0; off >>= 1) {
        if (t < off) red[t] += red[t + off];
        __syncthreads();
    }
    float tot = red[0] + 1e-16f;
#pragma unroll
    for (int j = 0; j < NN / 256; ++j)
        wout[(size_t)b * NN + t + j * 256] = wloc[j] / tot;
}

// ---------------------------------------------------------------------------
// mem_new = memory * (1 - ww*e) + ww*a ; float4 streaming (268MB R/W)
// ---------------------------------------------------------------------------
__global__ void mem_update(const float* __restrict__ mem, const float* __restrict__ ww,
                           const float* __restrict__ e, const float* __restrict__ a,
                           float* __restrict__ mem_new) {
    size_t idx = (size_t)blockIdx.x * blockDim.x + threadIdx.x;  // B*N*16 float4s
    if (idx >= (size_t)BB * NN * (MM / 4)) return;
    int m4 = (int)(idx & (MM / 4 - 1));
    size_t bn = idx >> 4;
    int n = (int)(bn & (NN - 1));
    int b = (int)(bn >> 11);
    float w = ww[(size_t)b * NN + n];
    float4 mv = ((const float4*)mem)[idx];
    float4 ev = ((const float4*)e)[(size_t)b * (MM / 4) + m4];
    float4 av = ((const float4*)a)[(size_t)b * (MM / 4) + m4];
    float4 o;
    o.x = mv.x * (1.0f - w * ev.x) + w * av.x;
    o.y = mv.y * (1.0f - w * ev.y) + w * av.y;
    o.z = mv.z * (1.0f - w * ev.z) + w * av.z;
    o.w = mv.w * (1.0f - w * ev.w) + w * av.w;
    ((float4*)mem_new)[idx] = o;
}

// ---------------------------------------------------------------------------
// read_new[b,m] = sum_n wr[b,n] * mem_new[b,n,m] ; lane-per-m coalesced
// ---------------------------------------------------------------------------
__global__ void read_dot(const float* __restrict__ mem_new,
                         const float* __restrict__ wr,
                         float* __restrict__ read_new) {
    int b = blockIdx.x;
    int m = threadIdx.x;                                   // 64 threads (2 waves)
    const float* Mb = mem_new + (size_t)b * NN * MM;
    const float* wb = wr + (size_t)b * NN;
    float acc = 0.0f;
    for (int n = 0; n < NN; ++n)
        acc += wb[n] * Mb[(size_t)n * MM + m];
    read_new[b * MM + m] = acc;
}

// ---------------------------------------------------------------------------
// o = sigmoid([h_new, read_new] @ Wo^T + bo)  (tiny: 2048 dots of K=320)
// ---------------------------------------------------------------------------
__global__ void out_head(const float* __restrict__ h, const float* __restrict__ rd,
                         const float* __restrict__ Wo, const float* __restrict__ bo,
                         float* __restrict__ o) {
    int idx = blockIdx.x * blockDim.x + threadIdx.x;       // B*NO
    if (idx >= BB * NO) return;
    int b = idx >> 3, j = idx & 7;
    const float* w = Wo + (size_t)j * (HH + MM);
    float acc = bo[j];
    for (int t = 0; t < HH; ++t) acc += h[(size_t)b * HH + t] * w[t];
    for (int m = 0; m < MM; ++m) acc += rd[(size_t)b * MM + m] * w[HH + m];
    o[idx] = sigf(acc);
}

// ---------------------------------------------------------------------------
// Host-side orchestration
// ---------------------------------------------------------------------------
extern "C" void kernel_launch(void* const* d_in, const int* in_sizes, int n_in,
                              void* d_out, int out_size, void* d_ws, size_t ws_size,
                              hipStream_t stream) {
    const float* x       = (const float*)d_in[0];
    const float* rd      = (const float*)d_in[1];
    const float* h       = (const float*)d_in[2];
    const float* c       = (const float*)d_in[3];
    const float* w_read  = (const float*)d_in[4];
    const float* w_write = (const float*)d_in[5];
    const float* memory  = (const float*)d_in[6];
    const float* W_ih    = (const float*)d_in[7];
    const float* W_hh    = (const float*)d_in[8];
    const float* b_ih    = (const float*)d_in[9];
    const float* b_hh    = (const float*)d_in[10];
    const float* Ww      = (const float*)d_in[11];
    const float* bw      = (const float*)d_in[12];
    const float* Wr      = (const float*)d_in[13];
    const float* br      = (const float*)d_in[14];
    const float* Wo      = (const float*)d_in[15];
    const float* bo      = (const float*)d_in[16];

    float* out = (float*)d_out;
    // flat output offsets (return order): o, read_new, h_new, c_new, wr, ww, mem_new
    float* o_o    = out;
    float* o_read = out + 2048;
    float* o_h    = out + 18432;
    float* o_c    = out + 83968;
    float* o_wr   = out + 149504;
    float* o_ww   = out + 673792;
    float* o_mem  = out + 1198080;

    // workspace layout (all chunks 16B-aligned)
    char* wsb = (char*)d_ws;
    _Float16* inpcat_h = (_Float16*)wsb;                 wsb += (size_t)BB * KCAT * 2;     // 180224
    _Float16* Wcat_h   = (_Float16*)wsb;                 wsb += (size_t)G4H * KCAT * 2;    // 720896
    _Float16* Wwp_h    = (_Float16*)wsb;                 wsb += (size_t)208 * HH * 2;      // 106496
    _Float16* Wrp_h    = (_Float16*)wsb;                 wsb += (size_t)80 * HH * 2;       // 40960
    _Float16* h16      = (_Float16*)wsb;                 wsb += (size_t)BB * HH * 2;       // 131072
    float* biascat  = (float*)wsb;                       wsb += (size_t)G4H * 4;
    float* gates    = (float*)wsb;                       wsb += (size_t)BB * G4H * 4;
    float* wout     = (float*)wsb;                       wsb += (size_t)BB * 208 * 4;
    float* rout     = (float*)wsb;                       wsb += (size_t)BB * 80 * 4;
    float* sim      = (float*)wsb;                       wsb += (size_t)BB * NN * 4;
    float* params_w = (float*)wsb;                       wsb += (size_t)BB * 8 * 4;
    float* params_r = (float*)wsb;                       wsb += (size_t)BB * 8 * 4;
    float* e_buf    = (float*)wsb;                       wsb += (size_t)BB * MM * 4;
    float* a_buf    = (float*)wsb;                       wsb += (size_t)BB * MM * 4;

    // 1) pack f16 operands for the WMMA GEMMs
    pack_inp<<<(BB * KCAT + 255) / 256, 256, 0, stream>>>(x, rd, h, inpcat_h);
    pack_w<<<(G4H * KCAT + 255) / 256, 256, 0, stream>>>(W_ih, W_hh, Wcat_h);
    pack_f16<<<(208 * HH + 255) / 256, 256, 0, stream>>>(Ww, 198, HH, Wwp_h, 208);
    pack_f16<<<(80 * HH + 255) / 256, 256, 0, stream>>>(Wr, 70, HH, Wrp_h, 80);
    pack_bias<<<(G4H + 255) / 256, 256, 0, stream>>>(b_ih, b_hh, biascat);

    // 2) gates = inpcat @ Wcat^T + bias   (16x64 tiles = 1024 waves = 256 blocks)
    wmma_gemm_bias<<<256, 128, 0, stream>>>(inpcat_h, KCAT, Wcat_h, KCAT,
                                            biascat, gates, G4H, G4H, KCAT, 64);

    // 3) LSTM activations -> h_new, c_new (straight into d_out) + f16 copy
    lstm_act<<<(BB * HH + 255) / 256, 256, 0, stream>>>(gates, c, o_c, o_h, h16);

    // 4) head projections: wout = h_new @ Ww^T + bw ; rout = h_new @ Wr^T + br
    wmma_gemm_bias<<<52, 128, 0, stream>>>(h16, HH, Wwp_h, HH, bw,
                                           wout, 208, 198, HH, 13);
    wmma_gemm_bias<<<20, 128, 0, stream>>>(h16, HH, Wrp_h, HH, br,
                                           rout, 80, 70, HH, 5);

    // 5) head scalar params (+ e/a for write head)
    head_params<<<1, 256, 0, stream>>>(wout, 208, params_w, e_buf, a_buf, 1);
    head_params<<<1, 256, 0, stream>>>(rout, 80, params_r, e_buf, a_buf, 0);

    // 6) write addressing on `memory`
    sim_kernel<<<dim3(NN / 256, BB), 256, 0, stream>>>(memory, wout, 208, params_w, sim);
    address_finalize<<<BB, 256, 0, stream>>>(sim, params_w, w_write, o_ww);

    // 7) erase/add memory update -> mem_new
    mem_update<<<(unsigned)((size_t)BB * NN * (MM / 4) / 256), 256, 0, stream>>>(
        memory, o_ww, e_buf, a_buf, o_mem);

    // 8) read addressing on mem_new
    sim_kernel<<<dim3(NN / 256, BB), 256, 0, stream>>>(o_mem, rout, 80, params_r, sim);
    address_finalize<<<BB, 256, 0, stream>>>(sim, params_r, w_read, o_wr);

    // 9) read vector + output head
    read_dot<<<BB, MM, 0, stream>>>(o_mem, o_wr, o_read);
    out_head<<<(BB * NO + 255) / 256, 256, 0, stream>>>(o_h, o_read, Wo, bo, o_o);
}
